// SPRGCN_88648124990278
// MI455X (gfx1250) — compile-verified, matrix operand order checked by reading, not run on hardware
//
#include <hip/hip_runtime.h>
#include <hip/hip_bf16.h>

typedef __attribute__((ext_vector_type(16))) _Float16 v16h;
typedef __attribute__((ext_vector_type(8)))  float    v8f;

// ---------------------------------------------------------------------------
// Degree / normalization:  deg[n] = 1 (self loop) + in-degree;  dinv = rsqrt
// ---------------------------------------------------------------------------
__global__ void deg_init_kernel(float* __restrict__ deg, int N) {
    int i = blockIdx.x * blockDim.x + threadIdx.x;
    if (i < N) deg[i] = 1.0f;
}

__global__ void deg_edges_kernel(const int* __restrict__ dst, float* __restrict__ deg, int E) {
    int e = blockIdx.x * blockDim.x + threadIdx.x;
    if (e < E) atomicAdd(&deg[dst[e]], 1.0f);
}

__global__ void deg_rsqrt_kernel(float* __restrict__ deg, int N) {
    int i = blockIdx.x * blockDim.x + threadIdx.x;
    if (i < N) deg[i] = rsqrtf(deg[i]);   // deg >= 1 always
}

// ---------------------------------------------------------------------------
// WMMA GEMM:  T[N,128] = H[N,KDIM] @ W[KDIM,128]
//   GATHER=true : H row n is emb[x[n]] (layer 1, KDIM=64)
//   GATHER=false: H row n is h[n]      (layer 2, KDIM=128)
// Block = 256 threads = 8 waves; wave w computes rows [blk*128 + w*16, +16)
// W is staged transposed in LDS as f16 so B fragments are contiguous ds loads.
// ---------------------------------------------------------------------------
__device__ __forceinline__ int frag_kbase(int p, int half) {
    // 16-bit A/B fragment K pattern: vgpr p holds K = kb, kb+1
    int base = (p < 4) ? (p * 2) : (16 + (p - 4) * 2);
    return base + half * 8;
}

template<int KDIM, bool GATHER>
__global__ void gemm_wmma_kernel(const float* __restrict__ Hsrc,
                                 const int*   __restrict__ xidx,
                                 const float* __restrict__ W,   // [KDIM][128]
                                 float*       __restrict__ T,   // [N][128]
                                 int N)
{
    __shared__ _Float16 Wt[128 * KDIM];          // Wt[c*KDIM + k] = W[k][c]
    const int tid = threadIdx.x;
    for (int i = tid; i < 128 * KDIM; i += 256) {
        int c = i / KDIM, k = i - c * KDIM;
        Wt[i] = (_Float16)W[k * 128 + c];
    }
    __syncthreads();

    const int lane = tid & 31;
    const int wave = tid >> 5;                   // 0..7
    const int half = lane >> 4;                  // K-half selector
    const int mr   = lane & 15;                  // row within tile (A) / col (B,D)
    const long long rowbase = (long long)blockIdx.x * 128 + wave * 16;

    int node = (int)rowbase + mr;
    int srcrow;
    if (GATHER) {
        int nn = node < N ? node : N - 1;        // clamp (store is guarded)
        srcrow = xidx[nn];
    } else {
        srcrow = node < N ? node : N - 1;
    }
    const float* rowptr = Hsrc + (long long)srcrow * KDIM;

    v8f acc[8];
#pragma unroll
    for (int t = 0; t < 8; ++t)
#pragma unroll
        for (int j = 0; j < 8; ++j) acc[t][j] = 0.0f;

#pragma unroll
    for (int kt = 0; kt < KDIM / 32; ++kt) {
        // ---- A fragment: row mr, K-slice [kt*32, kt*32+32) in documented layout
        v16h afrag;
#pragma unroll
        for (int p = 0; p < 8; ++p) {
            int kb = kt * 32 + frag_kbase(p, half);
            float2 w2 = *(const float2*)(rowptr + kb);      // 8B-aligned
            afrag[2 * p]     = (_Float16)w2.x;
            afrag[2 * p + 1] = (_Float16)w2.y;
        }
        // ---- 8 N-tiles of 16 columns, reuse the A fragment
#pragma unroll
        for (int nt = 0; nt < 8; ++nt) {
            const _Float16* cptr = &Wt[(nt * 16 + mr) * KDIM];
            v16h bfrag;
#pragma unroll
            for (int p = 0; p < 8; ++p) {
                int kb = kt * 32 + frag_kbase(p, half);
                bfrag[2 * p]     = cptr[kb];
                bfrag[2 * p + 1] = cptr[kb + 1];
            }
            acc[nt] = __builtin_amdgcn_wmma_f32_16x16x32_f16(
                false, afrag, false, bfrag, (short)0, acc[nt], false, false);
        }
    }

    // ---- store D: vgpr r -> M = r + 8*half, col = nt*16 + mr
#pragma unroll
    for (int nt = 0; nt < 8; ++nt)
#pragma unroll
        for (int r = 0; r < 8; ++r) {
            long long m = rowbase + r + 8 * half;
            if (m < N) T[m * 128 + nt * 16 + mr] = acc[nt][r];
        }
}

// ---------------------------------------------------------------------------
// H[n][c] = bias[c] + T[n][c] * dinv[n]^2   (self-loop term + bias init)
// ---------------------------------------------------------------------------
__global__ void init_h_kernel(const float* __restrict__ T, const float* __restrict__ dinv,
                              const float* __restrict__ bias, float* __restrict__ H, int N)
{
    long long i = (long long)blockIdx.x * blockDim.x + threadIdx.x;
    if (i >= (long long)N * 128) return;
    int n = (int)(i >> 7);
    int c = (int)(i & 127);
    float di = dinv[n];
    H[i] = bias[c] + T[i] * di * di;
}

// ---------------------------------------------------------------------------
// Edge aggregation: one wave per edge, lane handles 4 columns (float4 + atomics)
// H[dst] += T[src] * dinv[src]*dinv[dst]
// ---------------------------------------------------------------------------
__global__ void edge_agg_kernel(const int* __restrict__ src, const int* __restrict__ dst,
                                const float* __restrict__ dinv, const float* __restrict__ T,
                                float* __restrict__ H, int E)
{
    long long gid = (long long)blockIdx.x * blockDim.x + threadIdx.x;
    int e    = (int)(gid >> 5);
    int lane = (int)(gid & 31);
    if (e >= E) return;
    int s = src[e];
    int d = dst[e];
    float w = dinv[s] * dinv[d];
    float4 v = *(const float4*)(T + (long long)s * 128 + lane * 4);
    float* out = H + (long long)d * 128 + lane * 4;
    atomicAdd(out + 0, v.x * w);
    atomicAdd(out + 1, v.y * w);
    atomicAdd(out + 2, v.z * w);
    atomicAdd(out + 3, v.w * w);
}

__global__ void relu_kernel(float* __restrict__ H, long long total) {
    long long i = (long long)blockIdx.x * blockDim.x + threadIdx.x;
    if (i < total) H[i] = fmaxf(H[i], 0.0f);
}

// ---------------------------------------------------------------------------
// Max pool per graph (post-ReLU values >= 0 -> int bit-pattern atomicMax valid)
// ---------------------------------------------------------------------------
__global__ void pool_init_kernel(float* __restrict__ G, long long total) {
    long long i = (long long)blockIdx.x * blockDim.x + threadIdx.x;
    if (i < total) G[i] = 0.0f;
}

__global__ void pool_max_kernel(const float* __restrict__ H, const int* __restrict__ batch,
                                float* __restrict__ G, int N)
{
    long long i = (long long)blockIdx.x * blockDim.x + threadIdx.x;
    if (i >= (long long)N * 128) return;
    int n = (int)(i >> 7);
    int c = (int)(i & 127);
    int b = batch[n];
    atomicMax((int*)(G + (long long)b * 128 + c), __float_as_int(H[i]));
}

// ---------------------------------------------------------------------------
// Head: out[g][c] = blin[c] + sum_k G[g][k] * Wlin[k][c]   (tiny: 2000x10x128)
// ---------------------------------------------------------------------------
__global__ void head_kernel(const float* __restrict__ G, const float* __restrict__ Wlin,
                            const float* __restrict__ blin, float* __restrict__ out, int NG)
{
    int i = blockIdx.x * blockDim.x + threadIdx.x;
    if (i >= NG * 10) return;
    int g = i / 10;
    int c = i - g * 10;
    float acc = blin[c];
    const float* grow = G + (long long)g * 128;
#pragma unroll 8
    for (int k = 0; k < 128; ++k) acc += grow[k] * Wlin[k * 10 + c];
    out[i] = acc;
}

// ---------------------------------------------------------------------------
extern "C" void kernel_launch(void* const* d_in, const int* in_sizes, int n_in,
                              void* d_out, int out_size, void* d_ws, size_t ws_size,
                              hipStream_t stream)
{
    const int*   x     = (const int*)d_in[0];
    const int*   eidx  = (const int*)d_in[1];
    const int*   batch = (const int*)d_in[2];
    const float* emb   = (const float*)d_in[4];
    const float* W1    = (const float*)d_in[5];
    const float* b1    = (const float*)d_in[6];
    const float* W2    = (const float*)d_in[7];
    const float* b2    = (const float*)d_in[8];
    const float* Wlin  = (const float*)d_in[9];
    const float* blin  = (const float*)d_in[10];
    float* out = (float*)d_out;

    const int N  = in_sizes[0];       // 100000 nodes
    const int E  = in_sizes[1] / 2;   // 600000 edges
    const int NG = out_size / 10;     // 2000 graphs
    const int* esrc = eidx;           // edge_index[0]
    const int* edst = eidx + E;       // edge_index[1]

    // workspace layout (floats)
    float* ws   = (float*)d_ws;
    size_t off  = 0;
    float* dinv = ws + off; off += (size_t)((N + 255) & ~255);
    float* T    = ws + off; off += (size_t)N * 128;   // linear-transform output
    float* H    = ws + off; off += (size_t)N * 128;   // aggregated features
    float* G    = ws + off;                            // pooled [NG,128]

    const long long NF = (long long)N * 128;
    const int blk = 256;
    const int gN    = (N + blk - 1) / blk;
    const int gE    = (E + blk - 1) / blk;
    const int gNF   = (int)((NF + blk - 1) / blk);
    const int gEdge = (int)(((long long)E * 32 + blk - 1) / blk);
    const int gGemm = (N + 127) / 128;
    const int gPool = (int)(((long long)NG * 128 + blk - 1) / blk);

    // 1) normalization coefficients
    deg_init_kernel <<<gN, blk, 0, stream>>>(dinv, N);
    deg_edges_kernel<<<gE, blk, 0, stream>>>(edst, dinv, E);
    deg_rsqrt_kernel<<<gN, blk, 0, stream>>>(dinv, N);

    // 2) layer 1: t = emb[x] @ W1 ; h = b1 + self-loop + edge scatter ; relu
    gemm_wmma_kernel<64, true><<<gGemm, blk, 0, stream>>>(emb, x, W1, T, N);
    init_h_kernel   <<<gNF,   blk, 0, stream>>>(T, dinv, b1, H, N);
    edge_agg_kernel <<<gEdge, blk, 0, stream>>>(esrc, edst, dinv, T, H, E);
    relu_kernel     <<<gNF,   blk, 0, stream>>>(H, NF);

    // 3) layer 2: t = h @ W2 ; same aggregation ; relu
    gemm_wmma_kernel<128, false><<<gGemm, blk, 0, stream>>>(H, nullptr, W2, T, N);
    init_h_kernel   <<<gNF,   blk, 0, stream>>>(T, dinv, b2, H, N);
    edge_agg_kernel <<<gEdge, blk, 0, stream>>>(esrc, edst, dinv, T, H, E);
    relu_kernel     <<<gNF,   blk, 0, stream>>>(H, NF);

    // 4) per-graph max pool + classifier head
    pool_init_kernel<<<gPool, blk, 0, stream>>>(G, (long long)NG * 128);
    pool_max_kernel <<<gNF,   blk, 0, stream>>>(H, batch, G, N);
    head_kernel     <<<(NG * 10 + blk - 1) / blk, blk, 0, stream>>>(G, Wlin, blin, out, NG);
}
